// CorticalRecurrentStack_2645699854780
// MI455X (gfx1250) — matrix operand order repaired and checked by reading, not compiled, not run on hardware
//
#include <hip/hip_runtime.h>
#include <hip/hip_bf16.h>

constexpr int T_STEPS = 1024;
constexpr int BATCH   = 32;
constexpr int DIM     = 512;
constexpr int LAYERS  = 4;
constexpr int NWG     = 32;    // persistent workgroups; each owns DIM/NWG = 16 columns
constexpr int COLS    = 16;
constexpr int NTHR    = 256;   // 8 wave32s

typedef __attribute__((ext_vector_type(16))) __bf16 v16bf;
typedef __attribute__((ext_vector_type(8)))  __bf16 bf8;
typedef __attribute__((ext_vector_type(8)))  float  v8f;
typedef __attribute__((ext_vector_type(4)))  float  f4;

struct Params {
  const float* x;
  const float* Wf; const float* Uf; const float* bfb;
  const float* Wc; const float* Uc; const float* bcb;
  float* y;          // [T,B,D] output
  float* st;         // [L,B,D] final states
  __bf16* wbf;       // [L][2(f,c)][D][2*D (W then U)] bf16 weight cache
  __bf16* xbf;       // [T,B,D] bf16 copy of x
  __bf16* psbf;      // [L][B][D] bf16 broadcast of per-layer state
  unsigned* counter; // grid barrier
  unsigned* gen;
};

__device__ __forceinline__ bf8 ldbf8(const __bf16* p) {
  return *reinterpret_cast<const bf8*>(p);
}
__device__ __forceinline__ bf8 cvtbf8(const float* p) {
  f4 a = *reinterpret_cast<const f4*>(p);
  f4 b = *reinterpret_cast<const f4*>(p + 4);
  bf8 r;
  r[0]=(__bf16)a[0]; r[1]=(__bf16)a[1]; r[2]=(__bf16)a[2]; r[3]=(__bf16)a[3];
  r[4]=(__bf16)b[0]; r[5]=(__bf16)b[1]; r[6]=(__bf16)b[2]; r[7]=(__bf16)b[3];
  return r;
}
__device__ __forceinline__ v16bf pack16(bf8 lo, bf8 hi) {
  v16bf v;
  *reinterpret_cast<bf8*>(&v) = lo;
  *(reinterpret_cast<bf8*>(&v) + 1) = hi;
  return v;
}
// A 16x32 bf16: lane holds row M=lane%16, 16 consecutive K starting at kA+16*hi
__device__ __forceinline__ v16bf ldA(const __bf16* row, int kA, int hi) {
  return pack16(ldbf8(row + kA + hi * 16), ldbf8(row + kA + hi * 16 + 8));
}
__device__ __forceinline__ v16bf ldA32(const float* row, int kA, int hi) {
  return pack16(cvtbf8(row + kA + hi * 16), cvtbf8(row + kA + hi * 16 + 8));
}
// B 32x16 bf16: lane holds col N=lane%16, 16 consecutive K starting at kB=kA+16*hi
__device__ __forceinline__ v16bf ldB(const __bf16* row, int kB) {
  return pack16(ldbf8(row + kB), ldbf8(row + kB + 8));
}
__device__ __forceinline__ v16bf ldB32(const float* row, int kB) {
  return pack16(cvtbf8(row + kB), cvtbf8(row + kB + 8));
}

// Sense-reversing grid barrier (agent scope). `phase` is monotonically increasing.
__device__ __forceinline__ void grid_sync(unsigned* counter, unsigned* gen, unsigned phase) {
  __threadfence();
  __syncthreads();
  if (threadIdx.x == 0) {
    unsigned prev = __hip_atomic_fetch_add(counter, 1u, __ATOMIC_ACQ_REL, __HIP_MEMORY_SCOPE_AGENT);
    if (prev == (unsigned)(NWG - 1)) {
      __hip_atomic_store(counter, 0u, __ATOMIC_RELAXED, __HIP_MEMORY_SCOPE_AGENT);
      __hip_atomic_fetch_add(gen, 1u, __ATOMIC_ACQ_REL, __HIP_MEMORY_SCOPE_AGENT);
    } else {
      while (__hip_atomic_load(gen, __ATOMIC_ACQUIRE, __HIP_MEMORY_SCOPE_AGENT) < phase) {
        __builtin_amdgcn_s_sleep(2);
      }
    }
  }
  __syncthreads();
}

template <bool CACHED>
__global__ __launch_bounds__(NTHR, 1)
void crs_persistent(Params p) {
  __shared__ __align__(32) float red[4][32 * 8];   // K-half reduction per (mt,fc) tile
  __shared__ __align__(32) float zcb[2][32 * 8];   // Zc exchange per mt tile
  __shared__ float pslds[LAYERS][BATCH][COLS];     // fp32 recurrent state (own columns)

  const int tid  = threadIdx.x;
  const int lane = tid & 31;
  const int wv   = tid >> 5;         // 0..7
  const int mt   = wv & 1;           // batch tile (rows 0-15 / 16-31)
  const int fc   = (wv >> 1) & 1;    // 0 = f-gate preact, 1 = c-candidate preact
  const int kh   = wv >> 2;          // K-half of each 512-wide part
  const int nl   = lane & 15;
  const int hi   = lane >> 4;
  const int nglob = blockIdx.x * COLS + nl;  // owned output column (B operand)
  const int arow  = mt * 16 + nl;            // batch row supplied for A operand

  // ---------------- setup ---------------------------------------------------------
  {
    unsigned* pz = reinterpret_cast<unsigned*>(p.psbf);
    const int nz = LAYERS * BATCH * DIM / 2;
    for (int i = blockIdx.x * NTHR + tid; i < nz; i += NWG * NTHR) pz[i] = 0u;

    if constexpr (CACHED) {
      // bf16 weight cache: [L][f|c][n][ W(512) || U(512) ]
      const int nch = LAYERS * 2 * DIM * (2 * DIM) / 8;
      for (int c = blockIdx.x * NTHR + tid; c < nch; c += NWG * NTHR) {
        const int kk  = (c * 8) & (2 * DIM - 1);
        const int row = c >> 7;                 // (c*8) / 1024
        const int l   = row >> 10;
        const int rem = row & 1023;
        const int fcw = rem >> 9;
        const int n   = rem & 511;
        const float* src =
            (fcw ? (kk < DIM ? p.Wc : p.Uc) : (kk < DIM ? p.Wf : p.Uf)) +
            (size_t)(l * DIM + n) * DIM + (kk & (DIM - 1));
        *reinterpret_cast<bf8*>(p.wbf + (size_t)c * 8) = cvtbf8(src);
      }
      // bf16 copy of x
      const int nxc = T_STEPS * BATCH * DIM / 8;
      for (int c = blockIdx.x * NTHR + tid; c < nxc; c += NWG * NTHR)
        *reinterpret_cast<bf8*>(p.xbf + (size_t)c * 8) = cvtbf8(p.x + (size_t)c * 8);
    }
    for (int i = tid; i < LAYERS * BATCH * COLS; i += NTHR)
      (&pslds[0][0][0])[i] = 0.0f;
  }
  grid_sync(p.counter, p.gen, 1u);
  unsigned phase = 2u;

  // hoisted per-lane base pointers (loop-invariant over t)
  const __bf16* psrow[LAYERS];
  const __bf16* wrow[LAYERS][2];
  const float*  wrow32[LAYERS][2];
  float bfv[LAYERS], bcv[LAYERS];
  #pragma unroll
  for (int l = 0; l < LAYERS; ++l) {
    psrow[l] = p.psbf + (size_t)(l * BATCH + arow) * DIM;
    if constexpr (CACHED) {
      const __bf16* base = p.wbf + (size_t)((l * 2 + fc) * DIM + nglob) * (2 * DIM);
      wrow[l][0] = base;
      wrow[l][1] = base + DIM;
    } else {
      wrow32[l][0] = (fc ? p.Wc : p.Wf) + (size_t)(l * DIM + nglob) * DIM;
      wrow32[l][1] = (fc ? p.Uc : p.Uf) + (size_t)(l * DIM + nglob) * DIM;
    }
    bfv[l] = p.bfb[l * DIM + nglob];
    bcv[l] = p.bcb[l * DIM + nglob];
  }
  const __bf16* xrow  = CACHED ? (p.xbf + (size_t)arow * DIM) : nullptr;
  const float*  xrowf = CACHED ? nullptr : (p.x + (size_t)arow * DIM);

  // ---------------- recurrent chain: 4096 dependent stages ------------------------
  for (int t = 0; t < T_STEPS; ++t) {
    const __bf16* hrow = xrow;   // bf16 h-source for current layer (CACHED path)
    #pragma unroll
    for (int l = 0; l < LAYERS; ++l) {
      v8f acc{};
      // part 0: h @ W^T
      #pragma unroll
      for (int j = 0; j < 8; ++j) {
        const int kA = kh * 256 + j * 32;
        const int kB = kA + hi * 16;
        v16bf A, Bv;
        if constexpr (CACHED) {
          A  = ldA(hrow, kA, hi);
          Bv = ldB(wrow[l][0], kB);
        } else {
          if (l == 0) A = ldA32(xrowf, kA, hi);   // folds after unroll
          else        A = ldA(hrow, kA, hi);
          Bv = ldB32(wrow32[l][0], kB);
        }
        acc = __builtin_amdgcn_wmma_f32_16x16x32_bf16(
            false, A, false, Bv, (short)0, acc, false, false);
      }
      // part 1: ps @ U^T
      #pragma unroll
      for (int j = 0; j < 8; ++j) {
        const int kA = kh * 256 + j * 32;
        const int kB = kA + hi * 16;
        v16bf A = ldA(psrow[l], kA, hi);
        v16bf Bv;
        if constexpr (CACHED) Bv = ldB(wrow[l][1], kB);
        else                  Bv = ldB32(wrow32[l][1], kB);
        acc = __builtin_amdgcn_wmma_f32_16x16x32_bf16(
            false, A, false, Bv, (short)0, acc, false, false);
      }

      // reduce the two K-halves per tile through LDS
      if (kh == 1) *reinterpret_cast<v8f*>(&red[wv - 4][lane * 8]) = acc;
      __syncthreads();
      if (kh == 0) acc += *reinterpret_cast<const v8f*>(&red[wv][lane * 8]);
      if (wv == 2 || wv == 3)
        *reinterpret_cast<v8f*>(&zcb[wv - 2][lane * 8]) = acc;   // Zc totals
      __syncthreads();

      // gate math + state update (waves 0,1 hold Zf totals; mt == wv here)
      if (wv < 2) {
        const v8f zc = *reinterpret_cast<const v8f*>(&zcb[wv][lane * 8]);
        #pragma unroll
        for (int r = 0; r < 8; ++r) {
          const int b = mt * 16 + hi * 8 + r;    // C layout: VGPR r -> M = r + 8*hi
          const float f  = 1.0f / (1.0f + __expf(-(acc[r] + bfv[l])));
          const float cc = tanhf(zc[r] + bcv[l]);
          const float po = pslds[l][b][nl];
          const float ns = f * po + (1.0f - f) * cc;
          pslds[l][b][nl] = ns;
          p.psbf[(size_t)(l * BATCH + b) * DIM + nglob] = (__bf16)ns;
          if (l == LAYERS - 1) p.y[((size_t)t * BATCH + b) * DIM + nglob] = ns;
        }
      }
      grid_sync(p.counter, p.gen, phase++);
      hrow = psrow[l];   // next layer's h = this layer's new broadcast state
    }
    if constexpr (CACHED) xrow  += BATCH * DIM;
    else                  xrowf += BATCH * DIM;
  }

  // final states (fp32, own columns)
  __syncthreads();
  for (int i = tid; i < LAYERS * BATCH * COLS; i += NTHR) {
    const int l   = i >> 9;
    const int rem = i & 511;
    const int b   = rem >> 4;
    const int n2  = rem & 15;
    p.st[(size_t)(l * BATCH + b) * DIM + blockIdx.x * COLS + n2] = pslds[l][b][n2];
  }
}

extern "C" void kernel_launch(void* const* d_in, const int* in_sizes, int n_in,
                              void* d_out, int out_size, void* d_ws, size_t ws_size,
                              hipStream_t stream) {
  (void)in_sizes; (void)n_in; (void)out_size;
  Params p;
  p.x   = (const float*)d_in[0];
  p.Wf  = (const float*)d_in[1];
  p.Uf  = (const float*)d_in[2];
  p.bfb = (const float*)d_in[3];
  p.Wc  = (const float*)d_in[4];
  p.Uc  = (const float*)d_in[5];
  p.bcb = (const float*)d_in[6];
  p.y   = (float*)d_out;
  p.st  = (float*)d_out + (size_t)T_STEPS * BATCH * DIM;

  char* ws = (char*)d_ws;
  p.counter = (unsigned*)(ws + 0);
  p.gen     = (unsigned*)(ws + 128);
  p.psbf    = (__bf16*)(ws + 256);
  const size_t ps_bytes  = (size_t)LAYERS * BATCH * DIM * sizeof(__bf16);   // 128 KiB
  const size_t wbf_off   = 256 + ((ps_bytes + 255) & ~(size_t)255);
  const size_t wbf_bytes = (size_t)LAYERS * 2 * DIM * (2 * DIM) * sizeof(__bf16); // 8 MiB
  const size_t xbf_off   = wbf_off + wbf_bytes;
  const size_t xbf_bytes = (size_t)T_STEPS * BATCH * DIM * sizeof(__bf16);  // 32 MiB
  const bool cached = (ws_size >= xbf_off + xbf_bytes);
  p.wbf = cached ? (__bf16*)(ws + wbf_off) : (__bf16*)nullptr;
  p.xbf = cached ? (__bf16*)(ws + xbf_off) : (__bf16*)nullptr;

  hipMemsetAsync(d_ws, 0, 256, stream);  // reset grid-barrier words every call
  if (cached)
    hipLaunchKernelGGL((crs_persistent<true>),  dim3(NWG), dim3(NTHR), 0, stream, p);
  else
    hipLaunchKernelGGL((crs_persistent<false>), dim3(NWG), dim3(NTHR), 0, stream, p);
}
